// SelfAttention_31782757990864
// MI455X (gfx1250) — matrix operand order repaired and checked by reading, not compiled, not run on hardware
//
#include <hip/hip_runtime.h>
#include <hip/hip_bf16.h>

// ---------------------------------------------------------------------------
// Self-attention block for MI455X (gfx1250, wave32, WMMA).
//   DIM=3072, NUM_HEADS=24, HEAD_DIM=128, B=1, L=2048, fp32 in/out.
// bf16 WMMA (v_wmma_f32_16x16x32_bf16) for all four matmuls, f32 accumulate.
// Data movement uses both CDNA5 movers:
//   - GEMM: global_load_async_to_lds_b128 (+ s_wait_asynccnt) A-tile staging
//   - Flash attention: TDM tensor_load_to_lds (+ s_wait_tensorcnt) K/V tiles
// ---------------------------------------------------------------------------

#define DIM   3072
#define NHEAD 24
#define HD    128
#define SEQ   2048

typedef __bf16 bf16_t;
typedef __bf16 v16bf  __attribute__((ext_vector_type(16)));
typedef __bf16 bf16x8 __attribute__((ext_vector_type(8)));
typedef __bf16 bf16x4 __attribute__((ext_vector_type(4)));
typedef float  v8f    __attribute__((ext_vector_type(8)));
typedef unsigned u32x4 __attribute__((ext_vector_type(4)));
typedef unsigned u32x8 __attribute__((ext_vector_type(8)));

// Build a 16-element bf16 WMMA operand fragment from two 16-byte loads.
__device__ __forceinline__ v16bf make_frag(const bf16_t* p0, const bf16_t* p1) {
    bf16x8 lo = *(const bf16x8*)p0;
    bf16x8 hi = *(const bf16x8*)p1;
    v16bf r;
#pragma unroll
    for (int e = 0; e < 8; ++e) { r[e] = lo[e]; r[e + 8] = hi[e]; }
    return r;
}

__device__ __forceinline__ v8f wmma_bf16(v16bf a, v16bf b, v8f c) {
    return __builtin_amdgcn_wmma_f32_16x16x32_bf16(
        /*neg_a=*/false, a, /*neg_b=*/false, b,
        /*c_mod=*/(short)0, c, /*reuse_a=*/false, /*reuse_b=*/false);
}

// CDNA5 async DMA: copy 16B global -> LDS without touching VGPR data path.
// GVS addressing: mem = SADDR(64b) + VADDR(i32 byte offset). Tracked by
// ASYNCcnt. lds_addr is the 32-bit LDS byte address (low 32 bits of the
// generic pointer to a __shared__ object).
__device__ __forceinline__ void async_b128(unsigned lds_addr, const void* base,
                                           int byte_off) {
    asm volatile("global_load_async_to_lds_b128 %0, %1, %2"
                 :: "v"(lds_addr), "v"(byte_off), "s"(base) : "memory");
}
#define WAIT_ASYNC(imm) asm volatile("s_wait_asynccnt " #imm ::: "memory")
#define WAIT_DS0()      asm volatile("s_wait_dscnt 0x0" ::: "memory")

__device__ __forceinline__ unsigned lds_addr_of(const void* p) {
    return (unsigned)(size_t)p;
}

// CDNA5 Tensor Data Mover: one descriptor-driven DMA of a 2D tile
// (tile_d1 rows x tile_d0 elements, 2-byte elements, row stride stride0)
// from global to LDS, rows packed contiguously. D# groups per ISA 8.3/8.4;
// 2D tensor -> 2 SGPR groups (VADDR2/3 null). Tracked by TENSORcnt.
__device__ __forceinline__ void tdm_load_2d(unsigned lds_addr, const void* gptr,
                                            unsigned tensor_d0, unsigned tensor_d1,
                                            unsigned tile_d0, unsigned tile_d1,
                                            unsigned stride0) {
    const unsigned long long ga = (unsigned long long)gptr;
    u32x4 g0;
    g0[0] = 1u;                                   // count=1, user descriptor
    g0[1] = lds_addr;                             // lds_addr       [63:32]
    g0[2] = (unsigned)ga;                         // global_addr    [95:64]
    g0[3] = (unsigned)((ga >> 32) & 0x1FFFFFFu)   // global_addr    [120:96]
          | (2u << 30);                           // type=2 (image) [127:126]
    u32x8 g1;
    g1[0] = 1u << 16;                             // data_size=1 (2B), no flags
    g1[1] = (tensor_d0 & 0xFFFFu) << 16;          // tensor_dim0[15:0]  [63:48]
    g1[2] = (tensor_d0 >> 16)                     // tensor_dim0[31:16] [79:64]
          | ((tensor_d1 & 0xFFFFu) << 16);        // tensor_dim1[15:0]  [95:80]
    g1[3] = (tensor_d1 >> 16)                     // tensor_dim1[31:16] [111:96]
          | (tile_d0 << 16);                      // tile_dim0          [127:112]
    g1[4] = tile_d1;                              // tile_dim1; tile_dim2=0
    g1[5] = stride0;                              // tensor_dim0_stride[31:0]
    g1[6] = 0;                                    // stride0 hi, dim1_stride lo
    g1[7] = 0;                                    // dim1_stride hi (2D: unused)
    asm volatile("tensor_load_to_lds %0, %1" :: "s"(g0), "s"(g1) : "memory");
}

// ---------------------------------------------------------------------------
// fp32 -> bf16 conversion (vectorized x4)
// ---------------------------------------------------------------------------
__global__ void f32_to_bf16_v4(const float4* __restrict__ in,
                               bf16x4* __restrict__ out, int n4) {
    int i = blockIdx.x * blockDim.x + threadIdx.x;
    int stride = gridDim.x * blockDim.x;
    for (; i < n4; i += stride) {
        float4 v = in[i];
        bf16x4 o = { (bf16_t)v.x, (bf16_t)v.y, (bf16_t)v.z, (bf16_t)v.w };
        out[i] = o;
    }
}

// ---------------------------------------------------------------------------
// GEMM (NT): C[M,N] = A[M,K] * B[N,K]^T (+bias), A/B bf16 K-major, C fp32.
// Block: 256 threads = 8 waves arranged 2(M) x 4(N); wave tile 32x32;
// block tile 64(M) x 128(N). A block-tile (64x32) is staged through
// double-buffered LDS with async DMA; B fragments are unique per wave and
// come straight from the (L2-resident) weight matrix. 4 WMMA / wave / K-step.
// ---------------------------------------------------------------------------
__global__ void __launch_bounds__(256)
gemm_bf16_nt(const bf16_t* __restrict__ A, const bf16_t* __restrict__ B,
             float* __restrict__ C, const float* __restrict__ bias,
             int M, int N, int K) {
    __shared__ __align__(16) bf16_t abuf[2][64][32];   // 2 x 4KB

    const int lane = threadIdx.x & 31;
    const int wave = threadIdx.x >> 5;
    const int mw = wave >> 2;          // 0..1
    const int nw = wave & 3;           // 0..3
    const int bm0 = blockIdx.y * 64;
    const int bn = blockIdx.x * 128 + nw * 32;
    const int ln = lane & 15;
    const int hi = lane >> 4;          // half-wave select

    // one async B128 per thread per K-chunk: 256 x 16B = 4KB tile
    auto issue_a = [&](int k0, int buf) {
        const int c = threadIdx.x;
        const int row = c >> 2, col = (c & 3) * 8;
        async_b128(lds_addr_of(&abuf[buf][row][col]), A,
                   ((bm0 + row) * K + k0 + col) * 2);
    };

    v8f acc[2][2];
#pragma unroll
    for (int i = 0; i < 2; ++i)
#pragma unroll
        for (int j = 0; j < 2; ++j) acc[i][j] = (v8f)(0.0f);

    issue_a(0, 0);
    for (int k0 = 0, it = 0; k0 < K; k0 += 32, ++it) {
        const int buf = it & 1;
        if (k0 + 32 < K) {
            issue_a(k0 + 32, buf ^ 1);   // prefetch next tile into other buffer
            WAIT_ASYNC(0x1);             // current tile done (1 outstanding)
        } else {
            WAIT_ASYNC(0x0);
        }
        __syncthreads();

        v16bf a[2], b[2];
#pragma unroll
        for (int i = 0; i < 2; ++i) {
            // A fragment from LDS: lane half holds 16 contiguous K values
            const bf16_t* pa = &abuf[buf][mw * 32 + i * 16 + ln][hi * 16];
            a[i] = make_frag(pa, pa + 8);
        }
#pragma unroll
        for (int j = 0; j < 2; ++j) {
            // B fragment: lane half holds 16 contiguous K values
            const bf16_t* pb = B + (size_t)(bn + j * 16 + ln) * K + k0 + hi * 16;
            b[j] = make_frag(pb, pb + 8);
        }
#pragma unroll
        for (int i = 0; i < 2; ++i)
#pragma unroll
            for (int j = 0; j < 2; ++j)
                acc[i][j] = wmma_bf16(a[i], b[j], acc[i][j]);
        __syncthreads();                 // protect buffer before re-fill
    }

#pragma unroll
    for (int i = 0; i < 2; ++i)
#pragma unroll
        for (int j = 0; j < 2; ++j) {
            const int col = bn + j * 16 + ln;
            const float bv = bias ? bias[col] : 0.0f;
#pragma unroll
            for (int r = 0; r < 8; ++r) {
                const int row = bm0 + mw * 32 + i * 16 + r + hi * 8;
                C[(size_t)row * N + col] = acc[i][j][r] + bv;
            }
        }
}

// ---------------------------------------------------------------------------
// Fused QKNorm (RMSNorm over head_dim) + RoPE; writes q,k as bf16 (H, L, D).
// One wave per (token, head, q|k) row of 128; 4 elems/lane; shfl reduction.
// ---------------------------------------------------------------------------
__global__ void __launch_bounds__(128)
qknorm_rope(const float* __restrict__ qkv, const float* __restrict__ pe,
            const float* __restrict__ q_scale, const float* __restrict__ k_scale,
            bf16_t* __restrict__ qb, bf16_t* __restrict__ kb) {
    const int wave = threadIdx.x >> 5;
    const int lane = threadIdx.x & 31;
    const int row  = blockIdx.x * 4 + wave;          // 0 .. L*48-1
    const int l    = row / 48;
    const int rem  = row % 48;
    const int sel  = rem / 24;                       // 0 = q, 1 = k
    const int h    = rem % 24;
    const int d    = lane * 4;

    const float4 v = *(const float4*)(qkv + (size_t)l * (3 * DIM) +
                                      sel * DIM + h * HD + d);
    float ss = v.x * v.x + v.y * v.y + v.z * v.z + v.w * v.w;
#pragma unroll
    for (int m = 1; m < 32; m <<= 1) ss += __shfl_xor(ss, m);
    const float rms = rsqrtf(ss * (1.0f / 128.0f) + 1e-6f);

    const float* sc = sel ? k_scale : q_scale;
    const float x0 = v.x * rms * sc[d + 0];
    const float x1 = v.y * rms * sc[d + 1];
    const float x2 = v.z * rms * sc[d + 2];
    const float x3 = v.w * rms * sc[d + 3];

    // pe flat index: l*256 + p*4 + j*2 + i ; out[j] = f[j,0]*x0 + f[j,1]*x1
    const float* pb = pe + (size_t)l * 256 + (d / 2) * 4;
    const float o0 = pb[0] * x0 + pb[1] * x1;
    const float o1 = pb[2] * x0 + pb[3] * x1;
    const float o2 = pb[4] * x2 + pb[5] * x3;
    const float o3 = pb[6] * x2 + pb[7] * x3;

    bf16_t* dst = (sel ? kb : qb) + ((size_t)h * SEQ + l) * HD + d;
    bf16x4 ov = { (bf16_t)o0, (bf16_t)o1, (bf16_t)o2, (bf16_t)o3 };
    *(bf16x4*)dst = ov;
}

// ---------------------------------------------------------------------------
// V transpose: qkv[l][2*DIM + h*128 + d] (fp32) -> vT[h][d][l] (bf16),
// LDS-tiled 32x32 so both sides stay coalesced.
// ---------------------------------------------------------------------------
__global__ void __launch_bounds__(256)
v_transpose(const float* __restrict__ qkv, bf16_t* __restrict__ vT) {
    __shared__ float tile[32][33];
    const int h  = blockIdx.z;
    const int d0 = blockIdx.y * 32;
    const int l0 = blockIdx.x * 32;
    const int tx = threadIdx.x, ty = threadIdx.y;
#pragma unroll
    for (int r = 0; r < 4; ++r) {
        const int lr = ty + r * 8;
        tile[lr][tx] = qkv[(size_t)(l0 + lr) * (3 * DIM) + 2 * DIM + h * HD + d0 + tx];
    }
    __syncthreads();
#pragma unroll
    for (int r = 0; r < 4; ++r) {
        const int dr = ty + r * 8;
        vT[((size_t)h * HD + d0 + dr) * SEQ + l0 + tx] = (bf16_t)tile[tx][dr];
    }
}

// ---------------------------------------------------------------------------
// Flash attention. Block = 8 waves = 128 queries of one head; all waves share
// the key loop. K (32x128) and V^T (128x32) bf16 tiles are staged ONCE per
// block in double-buffered LDS by the Tensor Data Mover: wave 0 issues one
// tensor_load_to_lds per matrix per tile and waits on TENSORcnt; the
// workgroup barrier publishes the tile to all waves. Per 32-key tile and
// wave: 8 WMMA (S = Q K^T), shfl-butterfly online softmax over the 16-lane
// C-matrix halves, P -> LDS bounce (C-layout -> A-layout, s_wait_dscnt),
// 8 WMMA (O += P V).
// ---------------------------------------------------------------------------
__global__ void __launch_bounds__(256)
flash_attn(const bf16_t* __restrict__ qb, const bf16_t* __restrict__ kb,
           const bf16_t* __restrict__ vT, bf16_t* __restrict__ attn) {
    __shared__ __align__(16) bf16_t kbuf[2][32][HD];   // 2 x 8KB, keys x d
    __shared__ __align__(16) bf16_t vbuf[2][HD][32];   // 2 x 8KB, d x keys
    __shared__ __align__(16) bf16_t pbuf[8][16][32];   // per-wave P tile

    const int wave = threadIdx.x >> 5;
    const int lane = threadIdx.x & 31;
    const int ln   = lane & 15;
    const int hi   = lane >> 4;
    const int h    = blockIdx.y;
    const int q0   = blockIdx.x * 128 + wave * 16;

    const bf16_t* qhead = qb + (size_t)h * SEQ * HD;
    const bf16_t* khead = kb + (size_t)h * SEQ * HD;
    const bf16_t* vhead = vT + (size_t)h * HD * SEQ;

    // TDM: one K/V tile = 2 tensor_load_to_lds (descriptors built in SGPRs).
    auto issue_kv = [&](int kt, int buf) {
        // K tile: 32 rows (keys) x 128 elem, row stride HD -> kbuf[buf]
        tdm_load_2d(lds_addr_of(&kbuf[buf][0][0]), khead + (size_t)kt * HD,
                    /*tensor_d0=*/HD, /*tensor_d1=*/SEQ,
                    /*tile_d0=*/HD, /*tile_d1=*/32, /*stride0=*/HD);
        // V^T tile: 128 rows (d) x 32 elem (keys), row stride SEQ -> vbuf[buf]
        tdm_load_2d(lds_addr_of(&vbuf[buf][0][0]), vhead + kt,
                    /*tensor_d0=*/SEQ, /*tensor_d1=*/HD,
                    /*tile_d0=*/32, /*tile_d1=*/HD, /*stride0=*/SEQ);
    };

    // Q tile: 16 queries x 128d as 4 A-fragments (resident all loop long)
    v16bf aq[4];
#pragma unroll
    for (int c = 0; c < 4; ++c) {
        const bf16_t* p = qhead + (size_t)(q0 + ln) * HD + c * 32 + hi * 16;
        aq[c] = make_frag(p, p + 8);
    }

    v8f o[8];
#pragma unroll
    for (int dt = 0; dt < 8; ++dt) o[dt] = (v8f)(0.0f);
    float mrow[8], lrow[8];
#pragma unroll
    for (int r = 0; r < 8; ++r) { mrow[r] = -1e30f; lrow[r] = 0.0f; }
    const float scale = 0.08838834764831845f;   // 1/sqrt(128)

    if (wave == 0) issue_kv(0, 0);
    for (int it = 0; it < SEQ / 32; ++it) {
        const int buf = it & 1;
        if (wave == 0) {
            if (it + 1 < SEQ / 32) {
                issue_kv((it + 1) * 32, buf ^ 1);   // prefetch next tile
                // TDM ops from one wave complete in order: <=2 outstanding
                // means the current tile has fully landed in LDS.
                __builtin_amdgcn_s_wait_tensorcnt(2);
            } else {
                __builtin_amdgcn_s_wait_tensorcnt(0);
            }
        }
        __syncthreads();   // publish tile to all waves

        // S = Q K^T for 32 keys (two 16-key C tiles), K operands from LDS
        v8f s0 = (v8f)(0.0f), s1 = (v8f)(0.0f);
#pragma unroll
        for (int c = 0; c < 4; ++c) {
            const bf16_t* p0 = &kbuf[buf][ln][c * 32 + hi * 16];
            s0 = wmma_bf16(aq[c], make_frag(p0, p0 + 8), s0);
            const bf16_t* p1 = &kbuf[buf][16 + ln][c * 32 + hi * 16];
            s1 = wmma_bf16(aq[c], make_frag(p1, p1 + 8), s1);
        }

        // Online softmax. Row m = r + 8*hi lives entirely in one 16-lane half.
        float alpha[8];
#pragma unroll
        for (int r = 0; r < 8; ++r) {
            const float a = s0[r] * scale;
            const float b = s1[r] * scale;
            float mx = fmaxf(a, b);
            mx = fmaxf(mx, __shfl_xor(mx, 1));
            mx = fmaxf(mx, __shfl_xor(mx, 2));
            mx = fmaxf(mx, __shfl_xor(mx, 4));
            mx = fmaxf(mx, __shfl_xor(mx, 8));
            const float mnew = fmaxf(mrow[r], mx);
            alpha[r] = __expf(mrow[r] - mnew);
            mrow[r] = mnew;
            const float p0v = __expf(a - mnew);
            const float p1v = __expf(b - mnew);
            float su = p0v + p1v;
            su += __shfl_xor(su, 1);
            su += __shfl_xor(su, 2);
            su += __shfl_xor(su, 4);
            su += __shfl_xor(su, 8);
            lrow[r] = lrow[r] * alpha[r] + su;
            // P -> LDS (C layout: lane = key, vgpr = row)
            const int m = r + hi * 8;
            pbuf[wave][m][ln]      = (bf16_t)p0v;
            pbuf[wave][m][16 + ln] = (bf16_t)p1v;
        }

        // Rescale running O by alpha (per-row, matches C layout)
#pragma unroll
        for (int dt = 0; dt < 8; ++dt)
#pragma unroll
            for (int r = 0; r < 8; ++r) o[dt][r] *= alpha[r];

        // Wave-local LDS ordering: P stores complete before A-layout reloads.
        WAIT_DS0();

        // P as A fragment (lane = query row, K = key)
        const bf16_t* pp = &pbuf[wave][ln][hi * 16];
        const v16bf ap = make_frag(pp, pp + 8);

        // O += P V, V operands from LDS (rows contiguous in key)
#pragma unroll
        for (int dt = 0; dt < 8; ++dt) {
            const bf16_t* pv = &vbuf[buf][dt * 16 + ln][hi * 16];
            o[dt] = wmma_bf16(ap, make_frag(pv, pv + 8), o[dt]);
        }
        __syncthreads();   // all waves done with this buffer before re-fill
    }

    // O /= l, emit bf16 (L, H*D) layout for the projection GEMM.
    float inv[8];
#pragma unroll
    for (int r = 0; r < 8; ++r) inv[r] = 1.0f / lrow[r];
#pragma unroll
    for (int dt = 0; dt < 8; ++dt)
#pragma unroll
        for (int r = 0; r < 8; ++r)
            attn[(size_t)(q0 + r + hi * 8) * DIM + h * HD + dt * 16 + ln] =
                (bf16_t)(o[dt][r] * inv[r]);
}

// ---------------------------------------------------------------------------
// Host-side launch
// ---------------------------------------------------------------------------
extern "C" void kernel_launch(void* const* d_in, const int* in_sizes, int n_in,
                              void* d_out, int out_size, void* d_ws, size_t ws_size,
                              hipStream_t stream) {
    (void)in_sizes; (void)n_in; (void)out_size; (void)ws_size;
    const float* x       = (const float*)d_in[0];
    const float* pe      = (const float*)d_in[1];
    const float* w_qkv   = (const float*)d_in[2];
    const float* w_proj  = (const float*)d_in[3];
    const float* b_proj  = (const float*)d_in[4];
    const float* q_scale = (const float*)d_in[5];
    const float* k_scale = (const float*)d_in[6];
    float* out = (float*)d_out;

    // Workspace carve-up (256B aligned)
    char* ws = (char*)d_ws;
    size_t off = 0;
    auto alloc = [&](size_t bytes) { void* p = ws + off; off = (off + bytes + 255) & ~(size_t)255; return p; };
    bf16_t* xb     = (bf16_t*)alloc((size_t)SEQ * DIM * 2);
    bf16_t* wqkvb  = (bf16_t*)alloc((size_t)3 * DIM * DIM * 2);
    bf16_t* wprojb = (bf16_t*)alloc((size_t)DIM * DIM * 2);
    float*  qkv    = (float*)alloc((size_t)SEQ * 3 * DIM * 4);
    bf16_t* qb     = (bf16_t*)alloc((size_t)NHEAD * SEQ * HD * 2);
    bf16_t* kb     = (bf16_t*)alloc((size_t)NHEAD * SEQ * HD * 2);
    bf16_t* vTb    = (bf16_t*)alloc((size_t)NHEAD * HD * SEQ * 2);
    bf16_t* attnb  = (bf16_t*)alloc((size_t)SEQ * DIM * 2);

    // 1) fp32 -> bf16 conversions
    {
        int n4 = SEQ * DIM / 4;
        f32_to_bf16_v4<<<(n4 + 255) / 256, 256, 0, stream>>>((const float4*)x, (bf16x4*)xb, n4);
        n4 = 3 * DIM * DIM / 4;
        f32_to_bf16_v4<<<(n4 + 255) / 256, 256, 0, stream>>>((const float4*)w_qkv, (bf16x4*)wqkvb, n4);
        n4 = DIM * DIM / 4;
        f32_to_bf16_v4<<<(n4 + 255) / 256, 256, 0, stream>>>((const float4*)w_proj, (bf16x4*)wprojb, n4);
    }

    // 2) QKV GEMM: qkv[2048, 9216] = xb * wqkvb^T
    {
        dim3 grid((3 * DIM) / 128, SEQ / 64);
        gemm_bf16_nt<<<grid, 256, 0, stream>>>(xb, wqkvb, qkv, nullptr, SEQ, 3 * DIM, DIM);
    }

    // 3) Fused RMSNorm + RoPE for q,k -> bf16 (H, L, D)
    qknorm_rope<<<(SEQ * 48) / 4, 128, 0, stream>>>(qkv, pe, q_scale, k_scale, qb, kb);

    // 4) V -> bf16 transposed (H, D, L)
    {
        dim3 grid(SEQ / 32, HD / 32, NHEAD);
        v_transpose<<<grid, dim3(32, 8), 0, stream>>>(qkv, vTb);
    }

    // 5) Flash attention -> attnb bf16 (L, H*D)
    {
        dim3 grid(SEQ / 128, NHEAD);
        flash_attn<<<grid, 256, 0, stream>>>(qb, kb, vTb, attnb);
    }

    // 6) Projection GEMM: out[2048, 3072] = attnb * wprojb^T + b_proj
    {
        dim3 grid(DIM / 128, SEQ / 64);
        gemm_bf16_nt<<<grid, 256, 0, stream>>>(attnb, wprojb, out, b_proj, SEQ, DIM, DIM);
    }
}
